// GAT_28123445854611
// MI455X (gfx1250) — compile-verified, matrix operand order checked by reading, not run on hardware
//
#include <hip/hip_runtime.h>
#include <hip/hip_bf16.h>

// ---------------------------------------------------------------------------
// 2-layer GAT for MI455X (gfx1250, wave32).
//  - X/W pre-converted fp32->bf16 once; GEMM inner loop is pure b128 loads +
//    v_wmma_f32_16x16x32_bf16 (4 WMMAs / A-fragment: each wave owns a 16x64
//    C strip, 4 waves/block share a 64-col weight tile in LDS).
//  - Weight tile staged global->LDS with CDNA5 async copy
//    (global_load_async_to_lds_b128 / s_wait_asynccnt, ASYNCcnt path).
//  - Edge phase: order-encoded atomicMax for segment max, f32 atomics for
//    segment sum + scatter aggregation (HBM/atomic bound; L2-resident).
// ---------------------------------------------------------------------------

typedef __attribute__((ext_vector_type(16))) __bf16 v16bf;
typedef __attribute__((ext_vector_type(8)))  __bf16 v8bf;
typedef __attribute__((ext_vector_type(8)))  float  v8f;

#define HEADS 8

// ---- monotone float <-> uint encoding so unsigned atomicMax == float max ----
__device__ __forceinline__ unsigned ordenc(float f) {
    unsigned u = __float_as_uint(f);
    return (u & 0x80000000u) ? ~u : (u | 0x80000000u);
}
__device__ __forceinline__ float orddec(unsigned u) {
    return __uint_as_float((u & 0x80000000u) ? (u ^ 0x80000000u) : ~u);
}

// ---------------------------------------------------------------------------
// fp32 -> bf16 elementwise pre-convert (n must be a multiple of 8; all our
// tensor sizes are). One b128 store per thread.
// ---------------------------------------------------------------------------
__global__ void k_cvt_bf16(const float* __restrict__ in,
                           __bf16* __restrict__ out, long long n)
{
    long long i = ((long long)blockIdx.x * blockDim.x + threadIdx.x) * 8;
    if (i >= n) return;
    v8bf v;
#pragma unroll
    for (int j = 0; j < 8; ++j) v[j] = (__bf16)in[i + j];
    *(v8bf*)(out + i) = v;
}

// ---------------------------------------------------------------------------
// GEMM: Hout[M, Nout] = Xbf[M, K] * Wbf[Nout, K]^T   (bf16 in, f32 accum)
// Block = 128 threads = 4 waves. Block stages a contiguous 64-row (output
// col) tile of Wbf into LDS via async copy; wave w computes the 16x64 strip
// at rows (blockIdx.x*4 + w)*16, i.e. 4 WMMA N-tiles sharing one A fragment.
// ---------------------------------------------------------------------------
__global__ void __launch_bounds__(128)
k_gemm_wmma_bf16(const __bf16* __restrict__ Xbf, const __bf16* __restrict__ Wbf,
                 float* __restrict__ Hout, int K, int Nout, int Mtiles)
{
    __shared__ __bf16 wlds[64 * 256];          // up to 32 KB of 320 KB/WGP
    const int tid   = threadIdx.x;
    const int tileN = blockIdx.y;

    // ---- async-stage 64 output columns of Wbf (contiguous 64*K bf16) ------
    {
        const unsigned ldsbase = (unsigned)(size_t)&wlds[0]; // low 32b = LDS addr
        const char* gsrc = (const char*)(Wbf + (size_t)tileN * 64 * K);
        const int   totb = 64 * K * 2;
        for (int b = tid * 16; b < totb; b += 128 * 16) {
            unsigned           laddr = ldsbase + (unsigned)b;
            unsigned long long gaddr = (unsigned long long)(gsrc + b);
            asm volatile("global_load_async_to_lds_b128 %0, %1, off"
                         :: "v"(laddr), "v"(gaddr) : "memory");
        }
        asm volatile("s_wait_asynccnt 0x0" ::: "memory");
    }
    __syncthreads();

    const int wave  = tid >> 5;
    const int lane  = tid & 31;
    const int mtile = blockIdx.x * 4 + wave;
    if (mtile >= Mtiles) return;               // wave-uniform: EXEC all-1s below

    const int mrow  = mtile * 16 + (lane & 15);
    const int khalf = (lane >> 4) << 3;        // 0 (lanes 0-15) or 8 (16-31)
    const int lc    = lane & 15;               // local output column in sub-tile

    v8f c[4] = {};
    for (int k0 = 0; k0 < K; k0 += 32) {
        // A fragment: ISA 16-bit 16x32 layout -> two 16B contiguous chunks
        const __bf16* xp = Xbf + (size_t)mrow * K + k0 + khalf;
        __builtin_prefetch(xp + 32, 0, 0);     // next K-block (global_prefetch_b8)
        v8bf alo = *(const v8bf*)xp;
        v8bf ahi = *(const v8bf*)(xp + 16);
        v16bf a = __builtin_shufflevector(alo, ahi, 0, 1, 2, 3, 4, 5, 6, 7,
                                                   8, 9, 10, 11, 12, 13, 14, 15);
#pragma unroll
        for (int nt = 0; nt < 4; ++nt) {       // 4 B fragments reuse A
            const __bf16* wp = wlds + (size_t)(nt * 16 + lc) * K + k0 + khalf;
            v8bf blo = *(const v8bf*)wp;       // ds_load_b128
            v8bf bhi = *(const v8bf*)(wp + 16);
            v16bf b = __builtin_shufflevector(blo, bhi, 0, 1, 2, 3, 4, 5, 6, 7,
                                                       8, 9, 10, 11, 12, 13, 14, 15);
            c[nt] = __builtin_amdgcn_wmma_f32_16x16x32_bf16(
                        false, a, false, b, (short)0, c[nt], false, false);
        }
    }

    const int rbase = (lane >> 4) << 3;        // C layout: VGPR r -> M = r (+8)
#pragma unroll
    for (int nt = 0; nt < 4; ++nt)
#pragma unroll
        for (int r = 0; r < 8; ++r)
            Hout[(size_t)(mtile * 16 + rbase + r) * Nout
                 + tileN * 64 + nt * 16 + lc] = c[nt][r];
}

// ---------------------------------------------------------------------------
// alpha_s/alpha_d per (node, head); also initialize segment max/sum state.
// ---------------------------------------------------------------------------
__global__ void k_alpha(const float* __restrict__ h,
                        const float* __restrict__ a_src,
                        const float* __restrict__ a_dst,
                        float* __restrict__ as, float* __restrict__ ad,
                        unsigned* __restrict__ mkey, float* __restrict__ ssum,
                        int n, int F)
{
    int i = blockIdx.x * blockDim.x + threadIdx.x;
    if (i >= n * HEADS) return;
    int nid = i >> 3, hd = i & 7;
    const float* hp  = h + (size_t)nid * HEADS * F + hd * F;
    const float* asp = a_src + hd * F;
    const float* adp = a_dst + hd * F;
    float s1 = 0.f, s2 = 0.f;
    for (int f = 0; f < F; ++f) { float v = hp[f]; s1 += v * asp[f]; s2 += v * adp[f]; }
    as[i] = s1; ad[i] = s2;
    mkey[i] = 0u;        // ordenc lowest -> identity for max (self-loop guarantees data)
    ssum[i] = 0.f;
}

__device__ __forceinline__ void edge_sd(const int* ei, int E, long long e,
                                        int& s, int& d)
{
    if (e < E) { s = ei[e]; d = ei[E + e]; }
    else       { s = d = (int)(e - E); }       // appended self-loops
}

// pass 1: e = leaky_relu(as[src]+ad[dst]); segment max via atomic on ordenc
__global__ void k_edge_max(const int* __restrict__ ei, int E, int n,
                           const float* __restrict__ as, const float* __restrict__ ad,
                           float* __restrict__ ework, unsigned* __restrict__ mkey)
{
    int t = blockIdx.x * blockDim.x + threadIdx.x;
    int tot = (E + n) * HEADS;
    if (t >= tot) return;
    int hd = t & 7; long long e = t >> 3;
    int s, d; edge_sd(ei, E, e, s, d);
    float v = as[s * HEADS + hd] + ad[d * HEADS + hd];
    v = v > 0.f ? v : 0.2f * v;                // leaky_relu, slope 0.2
    ework[t] = v;
    atomicMax(&mkey[d * HEADS + hd], ordenc(v));
}

// pass 2: p = exp(e - max[dst]); segment sum
__global__ void k_edge_exp(const int* __restrict__ ei, int E, int n,
                           float* __restrict__ ework,
                           const unsigned* __restrict__ mkey,
                           float* __restrict__ ssum)
{
    int t = blockIdx.x * blockDim.x + threadIdx.x;
    int tot = (E + n) * HEADS;
    if (t >= tot) return;
    int hd = t & 7; long long e = t >> 3;
    int s, d; edge_sd(ei, E, e, s, d);
    float p = __expf(ework[t] - orddec(mkey[d * HEADS + hd]));
    ework[t] = p;
    atomicAdd(&ssum[d * HEADS + hd], p);
}

// pass 3: acc[dst] += h[src] * (p / sum[dst])
__global__ void k_aggregate(const int* __restrict__ ei, int E, int n,
                            const float* __restrict__ h,
                            const float* __restrict__ ework,
                            const float* __restrict__ ssum,
                            float* __restrict__ acc, int F)
{
    int t = blockIdx.x * blockDim.x + threadIdx.x;
    int tot = (E + n) * HEADS;
    if (t >= tot) return;
    int hd = t & 7; long long e = t >> 3;
    int s, d; edge_sd(ei, E, e, s, d);
    float alpha = ework[t] / (ssum[d * HEADS + hd] + 1e-16f);
    const float* hs = h   + (size_t)s * HEADS * F + hd * F;
    float*       ap = acc + (size_t)d * HEADS * F + hd * F;
    for (int f = 0; f < F; ++f) atomicAdd(&ap[f], hs[f] * alpha);
}

// mean over heads + bias + relu; output bf16 (feeds layer-2 WMMA directly)
__global__ void k_mean_bias_relu_bf16(const float* __restrict__ acc,
                                      const float* __restrict__ b,
                                      __bf16* __restrict__ out, int n, int F)
{
    int i = blockIdx.x * blockDim.x + threadIdx.x;
    if (i >= n * F) return;
    int nid = i / F, f = i - nid * F;
    const float* ap = acc + (size_t)nid * HEADS * F;
    float t = 0.f;
    for (int hd = 0; hd < HEADS; ++hd) t += ap[hd * F + f];
    t = t * (1.0f / HEADS) + b[f];
    out[i] = (__bf16)(t > 0.f ? t : 0.f);
}

// layer-2 tail: mean over heads + bias + relu + row softmax over 32
__global__ void k_final_softmax32(const float* __restrict__ acc,
                                  const float* __restrict__ b,
                                  float* __restrict__ out, int n)
{
    int nid = blockIdx.x * blockDim.x + threadIdx.x;
    if (nid >= n) return;
    const float* ap = acc + (size_t)nid * HEADS * 32;
    float v[32], mx = -1e30f;
    for (int f = 0; f < 32; ++f) {
        float t = 0.f;
        for (int hd = 0; hd < HEADS; ++hd) t += ap[hd * 32 + f];
        t = t * (1.0f / HEADS) + b[f];
        t = t > 0.f ? t : 0.f;
        v[f] = t; mx = fmaxf(mx, t);
    }
    float s = 0.f;
    for (int f = 0; f < 32; ++f) { v[f] = __expf(v[f] - mx); s += v[f]; }
    float inv = 1.0f / s;
    for (int f = 0; f < 32; ++f) out[(size_t)nid * 32 + f] = v[f] * inv;
}

// ---------------------------------------------------------------------------
extern "C" void kernel_launch(void* const* d_in, const int* in_sizes, int n_in,
                              void* d_out, int out_size, void* d_ws, size_t ws_size,
                              hipStream_t stream)
{
    const float* x     = (const float*)d_in[0];
    const int*   ei    = (const int*)  d_in[1];   // [2, E], per harness int contract
    const float* W1    = (const float*)d_in[2];   // [512, 256]
    const float* asrc1 = (const float*)d_in[3];
    const float* adst1 = (const float*)d_in[4];
    const float* b1    = (const float*)d_in[5];
    const float* W2    = (const float*)d_in[6];   // [256, 64]
    const float* asrc2 = (const float*)d_in[7];
    const float* adst2 = (const float*)d_in[8];
    const float* b2    = (const float*)d_in[9];
    float* outp = (float*)d_out;

    const int N    = in_sizes[0] / 256;           // 50000
    const int E    = in_sizes[1] / 2;             // 800000
    const int Etot = E + N;                       // + self loops

    // workspace layout (256B-aligned carve-outs); buffers aliased across layers
    char*  base = (char*)d_ws;
    size_t off  = 0;
    auto alloc = [&](size_t bytes) {
        void* p = base + off; off = (off + bytes + 255) & ~(size_t)255; return p;
    };
    float*    h1    = (float*)   alloc((size_t)N * 512 * 4);   // h1; aliased as h2
    float*    acc   = (float*)   alloc((size_t)N * 512 * 4);   // acc1 / acc2
    float*    as    = (float*)   alloc((size_t)N * HEADS * 4);
    float*    ad    = (float*)   alloc((size_t)N * HEADS * 4);
    float*    ssum  = (float*)   alloc((size_t)N * HEADS * 4);
    unsigned* mkey  = (unsigned*)alloc((size_t)N * HEADS * 4);
    float*    ework = (float*)   alloc((size_t)Etot * HEADS * 4);
    __bf16*   xbf   = (__bf16*)  alloc((size_t)N * 256 * 2);
    __bf16*   o1bf  = (__bf16*)  alloc((size_t)N * 64 * 2);
    __bf16*   W1bf  = (__bf16*)  alloc((size_t)512 * 256 * 2);
    __bf16*   W2bf  = (__bf16*)  alloc((size_t)256 * 64 * 2);
    (void)ws_size; (void)n_in; (void)out_size;

    const int Mtiles = N / 16;                    // 3125
    const int gEdge  = (Etot * HEADS + 255) / 256;
    const int gNH    = (N * HEADS + 255) / 256;

    // ---- one-time bf16 pre-conversions ------------------------------------
    {
        long long nx = (long long)N * 256;
        k_cvt_bf16<<<(int)((nx / 8 + 255) / 256), 256, 0, stream>>>(x, xbf, nx);
        k_cvt_bf16<<<(512 * 256 / 8 + 255) / 256, 256, 0, stream>>>(W1, W1bf, 512 * 256);
        k_cvt_bf16<<<(256 * 64 / 8 + 255) / 256, 256, 0, stream>>>(W2, W2bf, 256 * 64);
    }

    // ---------------- layer 1 (F = 64, K = 256, Nout = 512) ----------------
    {
        dim3 grid((Mtiles + 3) / 4, 512 / 64);
        k_gemm_wmma_bf16<<<grid, 128, 0, stream>>>(xbf, W1bf, h1, 256, 512, Mtiles);
    }
    k_alpha<<<gNH, 256, 0, stream>>>(h1, asrc1, adst1, as, ad, mkey, ssum, N, 64);
    k_edge_max<<<gEdge, 256, 0, stream>>>(ei, E, N, as, ad, ework, mkey);
    k_edge_exp<<<gEdge, 256, 0, stream>>>(ei, E, N, ework, mkey, ssum);
    hipMemsetAsync(acc, 0, (size_t)N * 512 * sizeof(float), stream);
    k_aggregate<<<gEdge, 256, 0, stream>>>(ei, E, N, h1, ework, ssum, acc, 64);
    k_mean_bias_relu_bf16<<<(N * 64 + 255) / 256, 256, 0, stream>>>(acc, b1, o1bf, N, 64);

    // ---------------- layer 2 (F = 32, K = 64, Nout = 256) -----------------
    float* h2 = h1;                               // alias: h1 dead after aggregate1
    {
        dim3 grid((Mtiles + 3) / 4, 256 / 64);
        k_gemm_wmma_bf16<<<grid, 128, 0, stream>>>(o1bf, W2bf, h2, 64, 256, Mtiles);
    }
    k_alpha<<<gNH, 256, 0, stream>>>(h2, asrc2, adst2, as, ad, mkey, ssum, N, 32);
    k_edge_max<<<gEdge, 256, 0, stream>>>(ei, E, N, as, ad, ework, mkey);
    k_edge_exp<<<gEdge, 256, 0, stream>>>(ei, E, N, ework, mkey, ssum);
    hipMemsetAsync(acc, 0, (size_t)N * 256 * sizeof(float), stream);
    k_aggregate<<<gEdge, 256, 0, stream>>>(ei, E, N, h2, ework, ssum, acc, 32);
    k_final_softmax32<<<(N + 127) / 128, 128, 0, stream>>>(acc, b2, outp, N);
}
